// NonMaximumSuppress_57148834841200
// MI455X (gfx1250) — 8-bit matrix operand order repaired and checked by reading, compiled, not run on hardware
//
#include <hip/hip_runtime.h>
#include <stdint.h>

typedef __attribute__((ext_vector_type(8))) int v8i;

#define NANCH   8732
#define KTOP    512
#define NCAND   1024
#define NTHR    1024
#define NCLS    21

struct Params {
  const float* feat[6];
  const float* def[6];
  float* out;
};

__device__ __forceinline__ void anchor_level(int a, int& l, int& r, int& cnt) {
  if (a < 5776)      { l = 0; r = a;        cnt = 5776; }
  else if (a < 7942) { l = 1; r = a - 5776; cnt = 2166; }
  else if (a < 8542) { l = 2; r = a - 7942; cnt = 600;  }
  else if (a < 8692) { l = 3; r = a - 8542; cnt = 150;  }
  else if (a < 8728) { l = 4; r = a - 8692; cnt = 36;   }
  else               { l = 5; r = a - 8728; cnt = 4;    }
}

__global__ __launch_bounds__(NTHR, 1)
void ssd_decode_topk_nms(Params P) {
  extern __shared__ unsigned char smem[];
  unsigned long long* cand = (unsigned long long*)smem;
  unsigned* uconf = (unsigned*)(cand + NCAND);
  int* hist = (int*)(uconf + 8736);
  int* ctl  = hist + 256;
  float* sX1 = (float*)(ctl + 4);
  float* sY1 = sX1 + KTOP;
  float* sX2 = sY1 + KTOP;
  float* sY2 = sX2 + KTOP;
  float* sAR = sY2 + KTOP;
  int*   sVL = (int*)(sAR + KTOP);
  float* sBOX = (float*)(sVL + KTOP);
  float* sKP  = sBOX + KTOP * 4;

  const int b   = blockIdx.x;
  const int tid = threadIdx.x;

  for (int a = tid; a < NANCH; a += NTHR) {
    int l, r, cnt;
    anchor_level(a, l, r, cnt);
    const float* fp = P.feat[l] + ((size_t)b * cnt + r) * (4 + NCLS);
    __builtin_prefetch(fp + (size_t)NTHR * (4 + NCLS), 0, 0);
    float m = -3.4e38f, m20 = -3.4e38f;
    float lg[NCLS];
    #pragma unroll
    for (int c = 0; c < NCLS; ++c) { lg[c] = fp[4 + c]; m = fmaxf(m, lg[c]); }
    #pragma unroll
    for (int c = 0; c < NCLS - 1; ++c) m20 = fmaxf(m20, lg[c]);
    float z = 0.f;
    #pragma unroll
    for (int c = 0; c < NCLS; ++c) z += __expf(lg[c] - m);
    float conf = __expf(m20 - m) / z;
    unsigned u = __float_as_uint(conf);
    u = (u & 0x80000000u) ? ~u : (u | 0x80000000u);
    uconf[a] = u;
  }
  __syncthreads();

  for (int pass = 0; pass < 4; ++pass) {
    const int shift = 24 - 8 * pass;
    for (int i = tid; i < 256; i += NTHR) hist[i] = 0;
    if (tid == 0 && pass == 0) { ctl[0] = KTOP; ctl[1] = 0; ctl[2] = 0; }
    __syncthreads();
    const unsigned prefix = (unsigned)ctl[1];
    const unsigned mask = pass ? (0xFFFFFFFFu << (shift + 8)) : 0u;
    for (int a = tid; a < NANCH; a += NTHR) {
      unsigned u = uconf[a];
      if (((u ^ prefix) & mask) == 0) atomicAdd(&hist[(u >> shift) & 255], 1);
    }
    __syncthreads();
    if (tid == 0) {
      int rank = ctl[0];
      int cum = 0, d = 256;
      while (d > 0) { --d; cum += hist[d]; if (cum >= rank) break; }
      ctl[0] = rank - (cum - hist[d]);
      ctl[1] = (int)(prefix | ((unsigned)d << shift));
    }
    __syncthreads();
  }
  const unsigned T = (unsigned)ctl[1];

  for (int a = tid; a < NANCH; a += NTHR) {
    unsigned u = uconf[a];
    if (u >= T) {
      int pos = atomicAdd(&ctl[2], 1);
      if (pos < NCAND)
        cand[pos] = ((unsigned long long)u << 32) |
                    (unsigned long long)(0xFFFFFFFFu - (unsigned)a);
    }
  }
  __syncthreads();
  {
    int cnt = ctl[2]; if (cnt > NCAND) cnt = NCAND;
    for (int i = tid; i < NCAND; i += NTHR)
      if (i >= cnt) cand[i] = 0ull;
  }
  __syncthreads();
  for (int k = 2; k <= NCAND; k <<= 1) {
    for (int j = k >> 1; j > 0; j >>= 1) {
      const int i = tid;
      const int p = i ^ j;
      if (p > i) {
        unsigned long long a0 = cand[i], a1 = cand[p];
        const bool up = ((i & k) == 0);
        if (up ? (a0 < a1) : (a0 > a1)) { cand[i] = a1; cand[p] = a0; }
      }
      __syncthreads();
    }
  }

  if (tid < KTOP) {
    unsigned long long key = cand[tid];
    unsigned hi = (unsigned)(key >> 32);
    float conf = __uint_as_float(hi & 0x7FFFFFFFu);
    int a = (int)(0xFFFFFFFFu - (unsigned)(key & 0xFFFFFFFFull));
    int l, r, cnt;
    anchor_level(a, l, r, cnt);
    const float* fp = P.feat[l] + ((size_t)b * cnt + r) * (4 + NCLS);
    const float* dp = P.def[l] + (size_t)r * 4;
    float c0 = fp[0], c1 = fp[1], c2 = fp[2], c3 = fp[3];
    float d0 = dp[0], d1 = dp[1], d2 = dp[2], d3 = dp[3];
    float b0 = d0 + d3 * c0;
    float b1 = d1 + d2 * c1;
    float b2 = d2 * __expf(c2);
    float b3 = d3 * __expf(c3);
    float x1 = b0 - b3 * 0.5f, y1 = b1 - b2 * 0.5f;
    float x2 = b0 + b3 * 0.5f, y2 = b1 + b2 * 0.5f;
    sX1[tid] = x1; sY1[tid] = y1; sX2[tid] = x2; sY2[tid] = y2;
    sAR[tid] = (x2 - x1) * (y2 - y1);
    sVL[tid] = (conf >= 0.5f) ? 1 : 0;
    sBOX[tid*4+0] = b0; sBOX[tid*4+1] = b1; sBOX[tid*4+2] = b2; sBOX[tid*4+3] = b3;
  }
  __syncthreads();

  {
    const int wave = tid >> 5;
    const int lane = tid & 31;
    const int j0 = wave * 16;
    v8i acc = {0,0,0,0,0,0,0,0};
    const v8i aones = {0x01010101,0x01010101,0x01010101,0x01010101,
                       0x01010101,0x01010101,0x01010101,0x01010101};
    for (int ic = 0; ic < 8; ++ic) {
      v8i bm;
      #pragma unroll
      for (int g = 0; g < 2; ++g) {
        const int j = j0 + (lane & 15);
        const float x1i = sX1[j], y1i = sY1[j], x2i = sX2[j], y2i = sY2[j];
        const float ari = sAR[j];
        #pragma unroll
        for (int vv = 0; vv < 4; ++vv) {
          unsigned w32 = 0;
          #pragma unroll
          for (int bb = 0; bb < 4; ++bb) {
            const int i = ic * 64 + (lane >> 4) * 32 + g * 16 + vv * 4 + bb;
            float lx = fmaxf(sX1[i], x1i);
            float ly = fmaxf(sY1[i], y1i);
            float rx = fminf(sX2[i], x2i);
            float ry = fminf(sY2[i], y2i);
            float iw = fmaxf(rx - lx, 0.f);
            float ih = fmaxf(ry - ly, 0.f);
            float inter = iw * ih;
            float iou = inter / (sAR[i] + ari - inter + 1e-9f);
            unsigned s = (sVL[i] && (i < j) && (iou >= 0.5f)) ? 1u : 0u;
            w32 |= s << (bb * 8);
          }
          bm[g * 4 + vv] = (int)w32;
        }
      }
      acc = __builtin_amdgcn_wmma_i32_16x16x64_iu8(false, aones, false, bm,
                                                   acc, false, false);
    }
    if (lane < 16) {
      const int j = j0 + lane;
      sKP[j] = (sVL[j] && acc[0] == 0) ? 1.0f : 0.0f;
    }
  }
  __syncthreads();

  if (tid < KTOP) {
    const float kp = sKP[tid];
    float* ob = P.out + ((size_t)b * KTOP + tid) * 4;
    ob[0] = sBOX[tid*4+0] * kp;
    ob[1] = sBOX[tid*4+1] * kp;
    ob[2] = sBOX[tid*4+2] * kp;
    ob[3] = sBOX[tid*4+3] * kp;
    P.out[(size_t)gridDim.x * KTOP * 4 + (size_t)b * KTOP + tid] = kp;
  }
}

extern "C" void kernel_launch(void* const* d_in, const int* in_sizes, int n_in,
                              void* d_out, int out_size, void* d_ws, size_t ws_size,
                              hipStream_t stream) {
  (void)n_in; (void)out_size; (void)d_ws; (void)ws_size;
  Params P;
  for (int i = 0; i < 6; ++i) {
    P.feat[i] = (const float*)d_in[2 * i];
    P.def[i]  = (const float*)d_in[2 * i + 1];
  }
  P.out = (float*)d_out;
  const int B = in_sizes[0] / (4 * 38 * 38 * 25);
  const size_t shmem = 66704;
  ssd_decode_topk_nms<<<dim3(B), dim3(NTHR), shmem, stream>>>(P);
}
